// Hmm_34746285425062
// MI455X (gfx1250) — matrix operand order repaired and checked by reading, not compiled, not run on hardware
//
#include <hip/hip_runtime.h>
#include <math.h>

// ---------------------------------------------------------------------------
// HMM forward (log-partition scan) for MI455X / gfx1250.
//
// Linear-space reformulation: per step t,
//   a = U_prev / S_prev            (== exp(alpha_{t-1}), normalized)
//   r_j = (a^T Texp)_j * exp(emit[j, x_t])
//   S_t = sum_j r_j ;  out[t] = log(S_t) ;  U = r
//
// Persistent cooperative kernel: 16 WGs, each holds a 64-column slice of
// Texp^T in LDS as bf16 and does its slice of the matvec with
// v_wmma_f32_16x16x32_bf16 (A = vector replicated over the 16 M rows).
// One deterministic grid barrier per step.
// ---------------------------------------------------------------------------

typedef __attribute__((ext_vector_type(16))) __bf16 v16bf;
typedef __attribute__((ext_vector_type(8)))  __bf16 bf16x8;
typedef __attribute__((ext_vector_type(8)))  float  v8f;

namespace {
constexpr int kK      = 1024;           // hidden states
constexpr int kNWG    = 16;             // workgroups (one column slice each)
constexpr int kCols   = kK / kNWG;      // 64 columns per WG
constexpr int kBlock  = 256;            // 8 wave32 per WG
constexpr int kStride = kK + 8;         // padded bf16 row stride -> conflict-free ds_load_b128
}

struct HmmWs {
  float U[2][kK];        // unnormalized alpha, double buffered
  float partial[kNWG];   // per-WG partial sums of r
  int   cnt;             // barrier arrival counter
  int   gen;             // barrier generation
};

__device__ inline void grid_barrier(int* cnt, int* gen, int phase) {
  __threadfence();               // make this thread's global stores visible
  __syncthreads();
  if (threadIdx.x == 0) {
    __builtin_amdgcn_s_cluster_barrier();   // NOP unless launched as a cluster
    if (__hip_atomic_fetch_add(cnt, 1, __ATOMIC_ACQ_REL,
                               __HIP_MEMORY_SCOPE_AGENT) == kNWG - 1) {
      __hip_atomic_store(cnt, 0, __ATOMIC_RELAXED, __HIP_MEMORY_SCOPE_AGENT);
      __hip_atomic_fetch_add(gen, 1, __ATOMIC_RELEASE, __HIP_MEMORY_SCOPE_AGENT);
    } else {
      while (__hip_atomic_load(gen, __ATOMIC_ACQUIRE,
                               __HIP_MEMORY_SCOPE_AGENT) < phase) {
        __builtin_amdgcn_s_sleep(1);
      }
    }
  }
  __syncthreads();
}

__global__ void __launch_bounds__(kBlock)
hmm_forward(const int* __restrict__ xs, const float* __restrict__ start,
            const float* __restrict__ trans, const float* __restrict__ emit,
            float* __restrict__ out, HmmWs* ws, int T, int V) {
  __shared__ __attribute__((aligned(16))) __bf16 Bmat[kCols * kStride]; // Texp^T slice
  __shared__ __attribute__((aligned(16))) __bf16 aLDS[kK];              // staged exp(alpha)
  __shared__ float redLDS[8 * kCols];                                   // cross-wave partials
  __shared__ float sred[2];
  __shared__ float sS;

  const int tid  = threadIdx.x;
  const int wg   = blockIdx.x;
  const int j0   = wg * kCols;          // first output column of this WG
  const int lane = tid & 31;
  const int wave = tid >> 5;

  // ---- One-time: gather + exp + transpose this WG's 64 columns of trans.
  // Bmat[c][k] = exp(trans[k, j0+c])  (bf16, k-contiguous for B fragments)
  for (int idx = tid; idx < kCols * kK; idx += kBlock) {
    int c = idx & (kCols - 1);
    int k = idx >> 6;
    Bmat[c * kStride + k] = (__bf16)__expf(trans[(size_t)k * kK + (j0 + c)]);
  }

  // ---- Step 0: U0[j] = exp(start[j] + emit[j, x0]); S0 = sum; out[0]=log S0
  int phase = 0;
  int x0 = xs[0];
  float r = 0.f;
  if (tid < kCols) {
    int j = j0 + tid;
    r = expf(start[j] + emit[(size_t)j * V + x0]);
    ws->U[0][j] = r;
  }
  __syncthreads();                       // Bmat also ready below this point
  float v = r;
  #pragma unroll
  for (int o = 16; o > 0; o >>= 1) v += __shfl_down(v, o, 32);
  if (lane == 0 && wave < 2) sred[wave] = v;
  __syncthreads();
  if (tid == 0) ws->partial[wg] = sred[0] + sred[1];
  ++phase;
  grid_barrier(&ws->cnt, &ws->gen, phase);
  if (tid == 0) {
    float s = 0.f;
    #pragma unroll
    for (int w = 0; w < kNWG; ++w) s += ws->partial[w];   // fixed order: deterministic
    sS = s;
    if (wg == 0) out[0] = logf(s);
  }
  __syncthreads();
  float invS = 1.0f / sS;

  const float* Ucur = ws->U[0];
  float*       Unxt = ws->U[1];

  const int halfA = (lane < 16) ? 0 : 8;    // A frag K-slice select (ISA 16-bit A 16x32)
  const int halfB = (lane < 16) ? 0 : 8;    // B frag K-slice select (ISA 16-bit B 32x16)
  const int ncol  = lane & 15;              // B/D column within an N-tile

  // ---- Steady-state scan: one WMMA matvec slice + one grid barrier per step.
  for (int t = 1; t < T; ++t) {
    // Emission gather for this step (issued early; consumed after matvec),
    // plus prefetch of next step's scattered emit reads (L2-resident table).
    int x = xs[t];
    float eraw = 0.f;
    if (tid < kCols) eraw = emit[(size_t)(j0 + tid) * V + x];
    if (t + 1 < T) {
      int xn = xs[t + 1];
      if (tid < kCols)
        __builtin_prefetch(&emit[(size_t)(j0 + tid) * V + xn], 0, 1);
    }

    // Stage normalized alpha into LDS as bf16.
    for (int i = tid; i < kK; i += kBlock)
      aLDS[i] = (__bf16)(Ucur[i] * invS);
    __syncthreads();

    // Wave `wave` covers K rows [wave*128, wave*128+128), all 4 N-tiles.
    v8f acc[4];
    #pragma unroll
    for (int nt = 0; nt < 4; ++nt) acc[nt] = (v8f)(0.0f);

    #pragma unroll
    for (int kt = 0; kt < 4; ++kt) {
      const int kb = wave * 128 + kt * 32;
      // A fragment (row-replicated vector): lane<16 -> K {kb..kb+7, kb+16..kb+23}
      //                                     lane>=16 -> K {kb+8..kb+15, kb+24..kb+31}
      bf16x8 alo = *(const bf16x8*)&aLDS[kb + halfA];
      bf16x8 ahi = *(const bf16x8*)&aLDS[kb + 16 + halfA];
      v16bf af = __builtin_shufflevector(alo, ahi, 0, 1, 2, 3, 4, 5, 6, 7, 8, 9,
                                         10, 11, 12, 13, 14, 15);
      #pragma unroll
      for (int nt = 0; nt < 4; ++nt) {
        // B fragment: lanes 0-15 hold K {kb..kb+7, kb+16..kb+23}, lanes 16-31 K {kb+8..kb+15, kb+24..kb+31}
        const __bf16* bp = &Bmat[(nt * 16 + ncol) * kStride + kb + halfB];
        bf16x8 blo = *(const bf16x8*)bp;
        bf16x8 bhi = *(const bf16x8*)(bp + 16);
        v16bf bfrag = __builtin_shufflevector(blo, bhi, 0, 1, 2, 3, 4, 5, 6, 7,
                                              8, 9, 10, 11, 12, 13, 14, 15);
        acc[nt] = __builtin_amdgcn_wmma_f32_16x16x32_bf16(
            false, af, false, bfrag, (short)0, acc[nt], false, false);
      }
    }

    // D row 0 (all rows replicated): acc[nt] element 0, lanes 0..15 = N 0..15.
    if (lane < 16) {
      #pragma unroll
      for (int nt = 0; nt < 4; ++nt)
        redLDS[wave * kCols + nt * 16 + lane] = acc[nt][0];
    }
    __syncthreads();

    float rr = 0.f;
    if (tid < kCols) {
      float b = 0.f;
      #pragma unroll
      for (int w = 0; w < 8; ++w) b += redLDS[w * kCols + tid];  // fixed order
      rr = b * expf(eraw);
      Unxt[j0 + tid] = rr;
    }
    float v2 = rr;
    #pragma unroll
    for (int o = 16; o > 0; o >>= 1) v2 += __shfl_down(v2, o, 32);
    if (lane == 0 && wave < 2) sred[wave] = v2;
    __syncthreads();
    if (tid == 0) ws->partial[wg] = sred[0] + sred[1];

    ++phase;
    grid_barrier(&ws->cnt, &ws->gen, phase);

    if (tid == 0) {
      float s = 0.f;
      #pragma unroll
      for (int w = 0; w < kNWG; ++w) s += ws->partial[w];
      sS = s;
      if (wg == 0) out[t] = logf(s);
    }
    __syncthreads();
    invS = 1.0f / sS;

    // Swap double buffers (safe: all reads of Ucur happened before the barrier).
    float* tmp = (float*)Ucur; Ucur = Unxt; Unxt = tmp;
  }
}

__global__ void hmm_init(HmmWs* ws) {
  ws->cnt = 0;
  ws->gen = 0;
}

extern "C" void kernel_launch(void* const* d_in, const int* in_sizes, int n_in,
                              void* d_out, int out_size, void* d_ws, size_t ws_size,
                              hipStream_t stream) {
  const int*   xs    = (const int*)d_in[0];
  const float* start = (const float*)d_in[1];
  const float* trans = (const float*)d_in[2];
  const float* emit  = (const float*)d_in[3];
  float*       out   = (float*)d_out;

  const int T = in_sizes[0];           // 4096
  const int K = in_sizes[1];           // 1024 (kernel assumes kK)
  const int V = in_sizes[3] / K;       // 32000
  (void)n_in; (void)out_size; (void)ws_size;

  HmmWs* ws = (HmmWs*)d_ws;            // needs ~8.2 KB of scratch
  hmm_init<<<1, 1, 0, stream>>>(ws);
  hmm_forward<<<kNWG, kBlock, 0, stream>>>(xs, start, trans, emit, out, ws, T, V);
}
